// NanoLLM_62294205661882
// MI455X (gfx1250) — 16-bit and 8-bit matrix operand order repaired and checked by reading, compiled, not run on hardware
//
#include <hip/hip_runtime.h>
#include <hip/hip_bf16.h>
#include <stdint.h>

#define VOCAB 32000
#define D     1024
#define H     16
#define DH    64
#define DFF   4096
#define NL    4
#define SEQ   2048
#define EPSI  1e-5f

typedef __attribute__((ext_vector_type(16))) _Float16 v16h;
typedef __attribute__((ext_vector_type(8)))  float    v8f;
typedef __attribute__((ext_vector_type(8)))  int      v8i;

union FragH { v16h h; uint4 q[2]; };
union FragI { v8i  v; uint4 q[2]; uint2 d[4]; };

__device__ __forceinline__ void async_g2l_b128(uint32_t lds_off, const void* gptr) {
  asm volatile("global_load_async_to_lds_b128 %0, %1, off"
               :: "v"(lds_off), "v"((unsigned long long)(uintptr_t)gptr)
               : "memory");
}
__device__ __forceinline__ void wait_async0() {
  asm volatile("s_wait_asynccnt 0x0" ::: "memory");
}
__device__ __forceinline__ void stage_tile(const uint8_t* gbase, size_t row_stride_b,
                                           int k0_bytes, void* lds_buf, int tid) {
  int row = tid >> 2, seg = tid & 3;
  const uint8_t* g = gbase + (size_t)row * row_stride_b + k0_bytes + seg * 16;
  uint32_t l = (uint32_t)(size_t)lds_buf + row * 64 + seg * 16;
  async_g2l_b128(l, g);
}

__device__ __forceinline__ float blockSum(float v, float* sh) {
  int t = threadIdx.x;
  sh[t] = v; __syncthreads();
  for (int s = blockDim.x >> 1; s > 0; s >>= 1) { if (t < s) sh[t] += sh[t + s]; __syncthreads(); }
  float r = sh[0]; __syncthreads(); return r;
}
__device__ __forceinline__ float blockMax(float v, float* sh) {
  int t = threadIdx.x;
  sh[t] = v; __syncthreads();
  for (int s = blockDim.x >> 1; s > 0; s >>= 1) { if (t < s) sh[t] = fmaxf(sh[t], sh[t + s]); __syncthreads(); }
  float r = sh[0]; __syncthreads(); return r;
}

__global__ void k_embed(const int* __restrict__ ids, const float* __restrict__ tok,
                        const float* __restrict__ pos, float* __restrict__ x) {
  int s = blockIdx.x;
  int id = ids[s];
  for (int i = threadIdx.x; i < D; i += blockDim.x)
    x[(size_t)s * D + i] = tok[(size_t)id * D + i] + pos[(size_t)s * D + i];
}

__global__ void k_abssum(const float* __restrict__ W, size_t n, float* __restrict__ accum) {
  __shared__ float sh[256];
  float s = 0.f;
  for (size_t i = blockIdx.x * (size_t)blockDim.x + threadIdx.x; i < n; i += (size_t)gridDim.x * blockDim.x)
    s += fabsf(W[i]);
  s = blockSum(s, sh);
  if (threadIdx.x == 0) atomicAdd(accum, s);
}

__global__ void k_quantw(const float* __restrict__ W, int8_t* __restrict__ Wq, size_t n,
                         const float* __restrict__ accum, float invn, float* __restrict__ gamma_out) {
  float gamma = accum[0] * invn;
  if (blockIdx.x == 0 && threadIdx.x == 0) gamma_out[0] = gamma;
  float inv = 1.f / (gamma + EPSI);
  for (size_t i = blockIdx.x * (size_t)blockDim.x + threadIdx.x; i < n; i += (size_t)gridDim.x * blockDim.x) {
    float q = rintf(W[i] * inv);
    q = fminf(1.f, fmaxf(-1.f, q));
    Wq[i] = (int8_t)q;
  }
}

__global__ void k_ln_quant(const float* __restrict__ x, const float* __restrict__ g,
                           int8_t* __restrict__ xq, float* __restrict__ sinv, int ldq) {
  __shared__ float sh[256];
  int row = blockIdx.x, tid = threadIdx.x;
  const float* xr = x + (size_t)row * D;
  float v[4], sum = 0.f;
#pragma unroll
  for (int j = 0; j < 4; ++j) { v[j] = xr[tid + j * 256]; sum += v[j]; }
  float mu = blockSum(sum, sh) * (1.f / D);
  float ss = 0.f;
#pragma unroll
  for (int j = 0; j < 4; ++j) { float d = v[j] - mu; ss += d * d; }
  float var = blockSum(ss, sh) * (1.f / D);
  float rst = rsqrtf(var + EPSI);
  float y[4], amax = 0.f;
#pragma unroll
  for (int j = 0; j < 4; ++j) { y[j] = (v[j] - mu) * rst * g[tid + j * 256]; amax = fmaxf(amax, fabsf(y[j])); }
  amax = fmaxf(blockMax(amax, sh), 1e-5f);
  float sc = 127.f / amax;
  if (tid == 0) sinv[row] = amax / 127.f;
#pragma unroll
  for (int j = 0; j < 4; ++j) {
    int q = (int)rintf(y[j] * sc);
    q = min(127, max(-128, q));
    xq[(size_t)row * ldq + tid + j * 256] = (int8_t)q;
  }
}

__global__ void k_absquant(const float* __restrict__ in, int8_t* __restrict__ xq,
                           float* __restrict__ sinv, int n, int relu2) {
  __shared__ float sh[256];
  int row = blockIdx.x, tid = threadIdx.x;
  const float* xr = in + (size_t)row * n;
  float amax = 0.f;
  for (int i = tid; i < n; i += 256) {
    float v = xr[i];
    if (relu2) { v = fmaxf(v, 0.f); v = v * v; }
    amax = fmaxf(amax, fabsf(v));
  }
  amax = fmaxf(blockMax(amax, sh), 1e-5f);
  float sc = 127.f / amax;
  if (tid == 0) sinv[row] = amax / 127.f;
  for (int i = tid; i < n; i += 256) {
    float v = xr[i];
    if (relu2) { v = fmaxf(v, 0.f); v = v * v; }
    int q = (int)rintf(v * sc);
    q = min(127, max(-128, q));
    xq[(size_t)row * n + i] = (int8_t)q;
  }
}

__global__ void k_ln_f16(const float* __restrict__ x, const float* __restrict__ g,
                         _Float16* __restrict__ out) {
  __shared__ float sh[256];
  int row = blockIdx.x, tid = threadIdx.x;
  const float* xr = x + (size_t)row * D;
  float v[4], sum = 0.f;
#pragma unroll
  for (int j = 0; j < 4; ++j) { v[j] = xr[tid + j * 256]; sum += v[j]; }
  float mu = blockSum(sum, sh) * (1.f / D);
  float ss = 0.f;
#pragma unroll
  for (int j = 0; j < 4; ++j) { float d = v[j] - mu; ss += d * d; }
  float var = blockSum(ss, sh) * (1.f / D);
  float rst = rsqrtf(var + EPSI);
#pragma unroll
  for (int j = 0; j < 4; ++j)
    out[(size_t)row * D + tid + j * 256] = (_Float16)((v[j] - mu) * rst * g[tid + j * 256]);
}

__global__ void k_f32tof16(const float* __restrict__ in, _Float16* __restrict__ out, size_t n) {
  for (size_t i = blockIdx.x * (size_t)blockDim.x + threadIdx.x; i < n; i += (size_t)gridDim.x * blockDim.x)
    out[i] = (_Float16)in[i];
}

__global__ void k_rope_pack(const float* __restrict__ qkv, _Float16* __restrict__ Qh,
                            _Float16* __restrict__ Kh, _Float16* __restrict__ Vt) {
  int s = blockIdx.x;
  int h = threadIdx.x >> 5, i = threadIdx.x & 31;
  const float* base = qkv + (size_t)s * 3 * D;
  float invf = powf(10000.f, -(float)i / 32.f);
  float ang = (float)s * invf;
  float c = cosf(ang), sn = sinf(ang);
  float q1 = base[h * 64 + i], q2 = base[h * 64 + i + 32];
  float k1 = base[D + h * 64 + i], k2 = base[D + h * 64 + i + 32];
  size_t o = ((size_t)h * SEQ + s) * 64;
  Qh[o + i]      = (_Float16)(q1 * c - q2 * sn);
  Qh[o + i + 32] = (_Float16)(q1 * sn + q2 * c);
  Kh[o + i]      = (_Float16)(k1 * c - k2 * sn);
  Kh[o + i + 32] = (_Float16)(k1 * sn + k2 * c);
  float v1 = base[2 * D + h * 64 + i], v2 = base[2 * D + h * 64 + i + 32];
  Vt[((size_t)h * 64 + i) * SEQ + s]      = (_Float16)v1;
  Vt[((size_t)h * 64 + i + 32) * SEQ + s] = (_Float16)v2;
}

__global__ void k_softmax(const float* __restrict__ scores, _Float16* __restrict__ probs) {
  __shared__ float sh[256];
  int r = blockIdx.x, tid = threadIdx.x;
  const float* sr = scores + (size_t)r * SEQ;
  float mx = -1e30f;
  for (int i = tid; i <= r; i += 256) mx = fmaxf(mx, sr[i]);
  mx = blockMax(mx, sh);
  float sum = 0.f;
  for (int i = tid; i <= r; i += 256) sum += __expf(sr[i] - mx);
  sum = blockSum(sum, sh);
  float inv = 1.f / sum;
  for (int i = tid; i < SEQ; i += 256) {
    float p = (i <= r) ? __expf(sr[i] - mx) * inv : 0.f;
    probs[(size_t)r * SEQ + i] = (_Float16)p;
  }
}

__global__ void k_gemm_f16nt(const _Float16* __restrict__ A, const _Float16* __restrict__ B,
                             float* __restrict__ C, int N, int K,
                             int lda, int ldb, int ldc, float alpha) {
  __shared__ uint8_t sB[2][4096];
  const int tid = threadIdx.x;
  const int wid = tid >> 5;
  const int lane = tid & 31;
  const int sel = lane >> 4;
  const int ln  = lane & 15;
  const int nb = N >> 6;
  const int bm = blockIdx.x / nb, bn = blockIdx.x % nb;

  const _Float16* arow = A + (size_t)(bm * 128 + wid * 16 + ln) * lda;
  const uint8_t*  gB   = (const uint8_t*)(B + (size_t)bn * 64 * ldb);
  const size_t    strB = (size_t)ldb * 2;

  v8f acc[4] = {};
  const int nch = K >> 5;
  stage_tile(gB, strB, 0, &sB[0][0], tid);
  for (int c = 0; c < nch; ++c) {
    const int cur = c & 1;
    wait_async0();
    __syncthreads();
    if (c + 1 < nch) stage_tile(gB, strB, (c + 1) * 64, &sB[cur ^ 1][0], tid);

    const int k0 = c * 32;
    FragH a;
    a.q[0] = *(const uint4*)(arow + k0 + 8 * sel);
    a.q[1] = *(const uint4*)(arow + k0 + 16 + 8 * sel);
    const uint8_t* bb = &sB[cur][0];
    FragH b[4];
#pragma unroll
    for (int j = 0; j < 4; ++j) {
      b[j].q[0] = *(const uint4*)(bb + (j * 16 + ln) * 64 + 16 * sel);
      b[j].q[1] = *(const uint4*)(bb + (j * 16 + ln) * 64 + 32 + 16 * sel);
    }
#pragma unroll
    for (int j = 0; j < 4; ++j)
      acc[j] = __builtin_amdgcn_wmma_f32_16x16x32_f16(false, a.h, false, b[j].h, (short)0,
                                                      acc[j], false, false);
  }
#pragma unroll
  for (int j = 0; j < 4; ++j)
#pragma unroll
    for (int v = 0; v < 8; ++v) {
      int m = bm * 128 + wid * 16 + v + 8 * sel;
      C[(size_t)m * ldc + bn * 64 + j * 16 + ln] = acc[j][v] * alpha;
    }
}

__global__ void k_gemm_i8nt(const int8_t* __restrict__ A, const int8_t* __restrict__ B,
                            const float* __restrict__ sinv, float* __restrict__ C,
                            int N, int K, int ldc, const float* __restrict__ gamma_p,
                            int residual) {
  __shared__ uint8_t sB[2][4096];
  const int tid = threadIdx.x;
  const int wid = tid >> 5;
  const int lane = tid & 31;
  const int sel = lane >> 4;
  const int ln  = lane & 15;
  const int nb = N >> 6;
  const int bm = blockIdx.x / nb, bn = blockIdx.x % nb;

  const int8_t*  arow = A + (size_t)(bm * 128 + wid * 16 + ln) * K;
  const uint8_t* gB   = (const uint8_t*)(B + (size_t)bn * 64 * K);

  v8i acc[4] = {};
  const int nch = K >> 6;
  stage_tile(gB, (size_t)K, 0, &sB[0][0], tid);
  for (int c = 0; c < nch; ++c) {
    const int cur = c & 1;
    wait_async0();
    __syncthreads();
    if (c + 1 < nch) stage_tile(gB, (size_t)K, (c + 1) * 64, &sB[cur ^ 1][0], tid);

    const int k0 = c * 64;
    FragI a;
    a.q[0] = *(const uint4*)(arow + k0 + 16 * sel);
    a.q[1] = *(const uint4*)(arow + k0 + 32 + 16 * sel);
    const uint8_t* bb = &sB[cur][0];
    FragI b[4];
#pragma unroll
    for (int j = 0; j < 4; ++j) {
      b[j].q[0] = *(const uint4*)(bb + (j * 16 + ln) * 64 + 16 * sel);
      b[j].q[1] = *(const uint4*)(bb + (j * 16 + ln) * 64 + 32 + 16 * sel);
    }
#pragma unroll
    for (int j = 0; j < 4; ++j)
      acc[j] = __builtin_amdgcn_wmma_i32_16x16x64_iu8(true, a.v, true, b[j].v, acc[j], false, false);
  }
  float gamma = gamma_p[0];
#pragma unroll
  for (int j = 0; j < 4; ++j)
#pragma unroll
    for (int v = 0; v < 8; ++v) {
      int m = bm * 128 + wid * 16 + v + 8 * sel;
      float dq = (float)acc[j][v] * gamma * sinv[m];
      size_t idx = (size_t)m * ldc + bn * 64 + j * 16 + ln;
      C[idx] = (residual ? C[idx] : 0.f) + dq;
    }
}

extern "C" void kernel_launch(void* const* d_in, const int* in_sizes, int n_in,
                              void* d_out, int out_size, void* d_ws, size_t ws_size,
                              hipStream_t stream) {
  const int*   ids  = (const int*)d_in[0];
  const float* tok  = (const float*)d_in[1];
  const float* pos  = (const float*)d_in[2];
  const float* Wqkv = (const float*)d_in[3];
  const float* Wo   = (const float*)d_in[4];
  const float* W1   = (const float*)d_in[5];
  const float* W2   = (const float*)d_in[6];
  const float* g1   = (const float*)d_in[7];
  const float* g2   = (const float*)d_in[8];
  const float* gf   = (const float*)d_in[9];

  uint8_t* p = (uint8_t*)d_ws;
  auto alloc = [&](size_t bytes) -> void* {
    void* r = (void*)p;
    p += (bytes + 255) & ~(size_t)255;
    return r;
  };
  float*    x        = (float*)alloc((size_t)SEQ * D * 4);
  float*    qkv      = (float*)alloc((size_t)SEQ * 3 * D * 4);
  float*    attn_out = (float*)alloc((size_t)SEQ * D * 4);
  float*    ffn1     = (float*)alloc((size_t)SEQ * DFF * 4);
  int8_t*   xq       = (int8_t*)alloc((size_t)SEQ * DFF);
  float*    sinv     = (float*)alloc((size_t)SEQ * 4);
  _Float16* Qh       = (_Float16*)alloc((size_t)H * SEQ * DH * 2);
  _Float16* Kh       = (_Float16*)alloc((size_t)H * SEQ * DH * 2);
  _Float16* Vt       = (_Float16*)alloc((size_t)H * DH * SEQ * 2);
  float*    scores   = (float*)alloc((size_t)SEQ * SEQ * 4);
  _Float16* probs    = (_Float16*)alloc((size_t)SEQ * SEQ * 2);
  _Float16* xh       = (_Float16*)alloc((size_t)SEQ * D * 2);
  _Float16* Wh       = (_Float16*)alloc((size_t)VOCAB * D * 2);
  int8_t*   Wq_qkv   = (int8_t*)alloc((size_t)NL * 3 * D * D);
  int8_t*   Wq_o     = (int8_t*)alloc((size_t)NL * D * D);
  int8_t*   Wq_1     = (int8_t*)alloc((size_t)NL * DFF * D);
  int8_t*   Wq_2     = (int8_t*)alloc((size_t)NL * D * DFF);
  float*    gacc     = (float*)alloc(16 * 4);
  float*    gammas   = (float*)alloc(16 * 4);

  hipMemsetAsync(gacc, 0, 16 * 4, stream);

  const float* wsrc[4] = {Wqkv, Wo, W1, W2};
  int8_t*      wdst[4] = {Wq_qkv, Wq_o, Wq_1, Wq_2};
  const size_t wn[4]   = {(size_t)3 * D * D, (size_t)D * D, (size_t)DFF * D, (size_t)D * DFF};
  for (int l = 0; l < NL; ++l) {
    for (int w = 0; w < 4; ++w) {
      int idx = l * 4 + w;
      const float* src = wsrc[w] + (size_t)l * wn[w];
      int8_t* dst = wdst[w] + (size_t)l * wn[w];
      k_abssum<<<1024, 256, 0, stream>>>(src, wn[w], gacc + idx);
      k_quantw<<<1024, 256, 0, stream>>>(src, dst, wn[w], gacc + idx,
                                         1.f / (float)wn[w], gammas + idx);
    }
  }

  k_embed<<<SEQ, 256, 0, stream>>>(ids, tok, pos, x);

  for (int l = 0; l < NL; ++l) {
    const float* gam = gammas + l * 4;
    k_ln_quant<<<SEQ, 256, 0, stream>>>(x, g1 + (size_t)l * D, xq, sinv, D);
    k_gemm_i8nt<<<(SEQ / 128) * (3 * D / 64), 256, 0, stream>>>(
        xq, Wq_qkv + (size_t)l * 3 * D * D, sinv, qkv, 3 * D, D, 3 * D, gam + 0, 0);
    k_rope_pack<<<SEQ, 512, 0, stream>>>(qkv, Qh, Kh, Vt);
    for (int h = 0; h < H; ++h) {
      k_gemm_f16nt<<<(SEQ / 128) * (SEQ / 64), 256, 0, stream>>>(
          Qh + (size_t)h * SEQ * DH, Kh + (size_t)h * SEQ * DH, scores,
          SEQ, DH, DH, DH, SEQ, 0.125f);
      k_softmax<<<SEQ, 256, 0, stream>>>(scores, probs);
      k_gemm_f16nt<<<(SEQ / 128) * (DH / 64), 256, 0, stream>>>(
          probs, Vt + (size_t)h * DH * SEQ, attn_out + h * DH,
          DH, SEQ, SEQ, SEQ, D, 1.0f);
    }
    k_absquant<<<SEQ, 256, 0, stream>>>(attn_out, xq, sinv, D, 0);
    k_gemm_i8nt<<<(SEQ / 128) * (D / 64), 256, 0, stream>>>(
        xq, Wq_o + (size_t)l * D * D, sinv, x, D, D, D, gam + 1, 1);
    k_ln_quant<<<SEQ, 256, 0, stream>>>(x, g2 + (size_t)l * D, xq, sinv, D);
    k_gemm_i8nt<<<(SEQ / 128) * (DFF / 64), 256, 0, stream>>>(
        xq, Wq_1 + (size_t)l * DFF * D, sinv, ffn1, DFF, D, DFF, gam + 2, 0);
    k_absquant<<<SEQ, 256, 0, stream>>>(ffn1, xq, sinv, DFF, 1);
    k_gemm_i8nt<<<(SEQ / 128) * (D / 64), 256, 0, stream>>>(
        xq, Wq_2 + (size_t)l * D * DFF, sinv, x, D, DFF, D, gam + 3, 1);
  }

  k_ln_f16<<<SEQ, 256, 0, stream>>>(x, gf, xh);
  k_f32tof16<<<4096, 256, 0, stream>>>(tok, Wh, (size_t)VOCAB * D);
  k_gemm_f16nt<<<(SEQ / 128) * (VOCAB / 64), 256, 0, stream>>>(
      xh, Wh, (float*)d_out, VOCAB, D, D, D, VOCAB, 1.0f);
}
